// Castle_41944650613176
// MI455X (gfx1250) — matrix operand order repaired and checked by reading, not compiled, not run on hardware
//
#include <hip/hip_runtime.h>
#include <stdint.h>

// ---------------------------------------------------------------------------
// Castle attention for MI455X (gfx1250, wave32, WMMA bf16 16x16x32).
// Pipeline: cvt(f32->bf16) -> QKV GEMM -> term1/lookahead tiles ->
//           fused flash (Su GEMM + Sc + silu + causal softmax + P@V) ->
//           output projection GEMM.
// This revision software-pipelines all WMMA K-loops (register double
// buffering) so fragment loads for step k+1 are in flight while step k's
// WMMAs execute, and widens the GEMM tiles to 32x64 per wave.
// ---------------------------------------------------------------------------

#define NSEQ 2048
#define NB   2
#define NH   8
#define DH   64
#define NBH  (NB * NH)

typedef __attribute__((ext_vector_type(16))) __bf16   v16bf;
typedef __attribute__((ext_vector_type(8)))  float    v8f;
typedef __attribute__((ext_vector_type(4)))  uint32_t u32x4;

union ABFrag { v16bf v; u32x4 q[2]; uint32_t w[8]; uint16_t h[16]; };

// Workspace layout (uint16_t elements)
static const size_t SLAB_E = (size_t)NB * NH * NSEQ * DH;        // 2,097,152 per qkv slab
static const size_t T1_E   = 6 * SLAB_E;                         // term1  [bh][N][N] bf16
static const size_t NN_E   = (size_t)NBH * NSEQ * NSEQ;          // 67,108,864
static const size_t LA_E   = T1_E + NN_E;                        // lookahead
static const size_t OH_E   = LA_E + NN_E;                        // out_heads [B,N,512] bf16
static const size_t XB_E   = OH_E + (size_t)NB * NSEQ * 512;     // x bf16 [4096,1024]
static const size_t WQ_E   = XB_E + (size_t)NB * NSEQ * 1024;    // W_qkv bf16 [3072,1024]
static const size_t WO_E   = WQ_E + (size_t)3072 * 1024;         // W_out bf16 [1024,512]

__device__ __forceinline__ uint16_t f2bf(float f) {
  uint32_t u = __float_as_uint(f);
  return (uint16_t)((u + 0x7FFFu + ((u >> 16) & 1u)) >> 16);
}

// A fragment: 16x32 bf16 (MxK). Lane holds row (caller passes row incl. lane&15);
// K chunks at k0 + 8*(lane>>4) and k0 + 16 + 8*(lane>>4): two b128 loads.
__device__ __forceinline__ v16bf load_a_frag(const uint16_t* base, int row, int ld,
                                             int k0, int lane) {
  ABFrag f;
  const int lh = lane >> 4;
  const uint16_t* p = base + (size_t)row * ld + k0 + 8 * lh;
  f.q[0] = *(const u32x4*)p;
  f.q[1] = *(const u32x4*)(p + 16);
  return f.v;
}

// B fragment: 32x16 bf16 (KxN). Lane holds column; K chunks at
// k0 + 8*(lane>>4) and k0 + 16 + 8*(lane>>4): two b128 loads.
__device__ __forceinline__ v16bf load_b_frag(const uint16_t* base, int col, int ld,
                                             int k0, int lane) {
  ABFrag f;
  const int lh = lane >> 4;
  const uint16_t* p = base + (size_t)col * ld + k0 + 8 * lh;
  f.q[0] = *(const u32x4*)p;
  f.q[1] = *(const u32x4*)(p + 16);
  return f.v;
}

__device__ __forceinline__ v8f wmma_bf16(v16bf a, v16bf b, v8f c) {
  return __builtin_amdgcn_wmma_f32_16x16x32_bf16(false, a, false, b, (short)0, c,
                                                 false, false);
}

// ---------------------------------------------------------------------------
// Kernel 0: f32 -> bf16 conversion
// ---------------------------------------------------------------------------
__global__ void cvt_kernel(const float* __restrict__ src, uint16_t* __restrict__ dst,
                           int n) {
  int i = blockIdx.x * blockDim.x + threadIdx.x;
  if (i < n) dst[i] = f2bf(src[i]);
}

// ---------------------------------------------------------------------------
// Kernel 1: QKV projection. C[4096,3072] = x_bf @ Wqkv_bf^T, scatter into
// six [B,h,N,64] slabs (vc stored transposed [B,h,64,N]); SCALE folded into
// qu (s=0) and qc (s=3). One wave = 32x64 tile, K=1024, double-buffered.
// ---------------------------------------------------------------------------
__global__ void __launch_bounds__(128)
qkv_gemm(const uint16_t* __restrict__ xb, const uint16_t* __restrict__ wq,
         uint16_t* __restrict__ qkv) {
  const int wid  = (blockIdx.x * 128 + threadIdx.x) >> 5;
  const int lane = threadIdx.x & 31;
  const int lh = lane >> 4, ln = lane & 15;
  const int mt = wid / 48, nt = wid % 48;
  const int m0 = mt * 32, n0 = nt * 64;

  v8f acc[2][4] = {};
  const int rA0 = m0 + ln, rA1 = m0 + 16 + ln;

  v16bf a0c = load_a_frag(xb, rA0, 1024, 0, lane);
  v16bf a1c = load_a_frag(xb, rA1, 1024, 0, lane);
  v16bf bc0 = load_b_frag(wq, n0 + 0 * 16 + ln, 1024, 0, lane);
  v16bf bc1 = load_b_frag(wq, n0 + 1 * 16 + ln, 1024, 0, lane);
  v16bf bc2 = load_b_frag(wq, n0 + 2 * 16 + ln, 1024, 0, lane);
  v16bf bc3 = load_b_frag(wq, n0 + 3 * 16 + ln, 1024, 0, lane);

  for (int k0 = 0; k0 < 1024; k0 += 32) {
    v16bf a0n = {}, a1n = {}, bn0 = {}, bn1 = {}, bn2 = {}, bn3 = {};
    const int kn = k0 + 32;
    if (kn < 1024) {   // prefetch next K-step while current WMMAs run
      a0n = load_a_frag(xb, rA0, 1024, kn, lane);
      a1n = load_a_frag(xb, rA1, 1024, kn, lane);
      bn0 = load_b_frag(wq, n0 + 0 * 16 + ln, 1024, kn, lane);
      bn1 = load_b_frag(wq, n0 + 1 * 16 + ln, 1024, kn, lane);
      bn2 = load_b_frag(wq, n0 + 2 * 16 + ln, 1024, kn, lane);
      bn3 = load_b_frag(wq, n0 + 3 * 16 + ln, 1024, kn, lane);
    }
    acc[0][0] = wmma_bf16(a0c, bc0, acc[0][0]);
    acc[0][1] = wmma_bf16(a0c, bc1, acc[0][1]);
    acc[0][2] = wmma_bf16(a0c, bc2, acc[0][2]);
    acc[0][3] = wmma_bf16(a0c, bc3, acc[0][3]);
    acc[1][0] = wmma_bf16(a1c, bc0, acc[1][0]);
    acc[1][1] = wmma_bf16(a1c, bc1, acc[1][1]);
    acc[1][2] = wmma_bf16(a1c, bc2, acc[1][2]);
    acc[1][3] = wmma_bf16(a1c, bc3, acc[1][3]);
    a0c = a0n; a1c = a1n; bc0 = bn0; bc1 = bn1; bc2 = bn2; bc3 = bn3;
  }
#pragma unroll
  for (int g = 0; g < 2; ++g) {
#pragma unroll
    for (int s = 0; s < 4; ++s) {
#pragma unroll
      for (int r = 0; r < 8; ++r) {
        const int m = m0 + g * 16 + r + 8 * lh;  // global row = b*N + n
        const int c = n0 + s * 16 + ln;          // 0..3071
        const int bb = m >> 11, nn = m & (NSEQ - 1);
        const int sq = c >> 9;                   // which of 6 projections
        const int hh = (c >> 6) & 7;
        const int dd = c & 63;
        float v = acc[g][s][r];
        if (sq == 0 || sq == 3) v *= 0.125f;     // SCALE = 64^-0.5
        const uint16_t bv = f2bf(v);
        const int bh = bb * NH + hh;
        if (sq == 5) {  // vc transposed: [bh][dd][n]
          qkv[5 * SLAB_E + ((size_t)bh * DH + dd) * NSEQ + nn] = bv;
        } else {        // [bh][n][dd]
          qkv[(size_t)sq * SLAB_E + ((size_t)bh * NSEQ + nn) * DH + dd] = bv;
        }
      }
    }
  }
}

// ---------------------------------------------------------------------------
// Kernel 2: term1 = mask_lower(qc_s @ vu^T), lookahead = strict_upper(
// sigmoid(qu_s @ ku^T)). One wave = 16x64 tile of both matrices, K=64 fully
// unrolled with all fragment loads hoisted ahead of the WMMAs.
// ---------------------------------------------------------------------------
__global__ void __launch_bounds__(128)
tl_kernel(const uint16_t* __restrict__ qu, const uint16_t* __restrict__ ku,
          const uint16_t* __restrict__ vu, const uint16_t* __restrict__ qc,
          uint16_t* __restrict__ t1, uint16_t* __restrict__ la) {
  const int wid  = (blockIdx.x * 128 + threadIdx.x) >> 5;
  const int lane = threadIdx.x & 31;
  const int lh = lane >> 4, ln = lane & 15;
  const int bh  = wid >> 12;          // 128 i-tiles * 32 j-tiles per (b,h)
  const int rem = wid & 4095;
  const int it = rem >> 5, jt = rem & 31;
  const int i0 = it * 16, j0 = jt * 64;

  const uint16_t* qc_b = qc + (size_t)bh * NSEQ * DH;
  const uint16_t* qu_b = qu + (size_t)bh * NSEQ * DH;
  const uint16_t* vu_b = vu + (size_t)bh * NSEQ * DH;
  const uint16_t* ku_b = ku + (size_t)bh * NSEQ * DH;

  const bool need_t1 = (j0 <= i0 + 15);        // any j <= i in tile
  const bool need_la = (j0 + 63 > i0);         // any j >  i in tile

  v8f accT[4] = {};
  v8f accL[4] = {};
  const int rowA = i0 + ln;
  if (need_t1) {
    const v16bf a0 = load_a_frag(qc_b, rowA, DH, 0, lane);
    const v16bf a1 = load_a_frag(qc_b, rowA, DH, 32, lane);
#pragma unroll
    for (int s = 0; s < 4; ++s) {
      const v16bf b0 = load_b_frag(vu_b, j0 + s * 16 + ln, DH, 0, lane);
      const v16bf b1 = load_b_frag(vu_b, j0 + s * 16 + ln, DH, 32, lane);
      accT[s] = wmma_bf16(a0, b0, accT[s]);
      accT[s] = wmma_bf16(a1, b1, accT[s]);
    }
  }
  if (need_la) {
    const v16bf a0 = load_a_frag(qu_b, rowA, DH, 0, lane);
    const v16bf a1 = load_a_frag(qu_b, rowA, DH, 32, lane);
#pragma unroll
    for (int s = 0; s < 4; ++s) {
      const v16bf b0 = load_b_frag(ku_b, j0 + s * 16 + ln, DH, 0, lane);
      const v16bf b1 = load_b_frag(ku_b, j0 + s * 16 + ln, DH, 32, lane);
      accL[s] = wmma_bf16(a0, b0, accL[s]);
      accL[s] = wmma_bf16(a1, b1, accL[s]);
    }
  }
  const size_t base = (size_t)bh * NSEQ * NSEQ;
#pragma unroll
  for (int s = 0; s < 4; ++s) {
#pragma unroll
    for (int r = 0; r < 8; ++r) {
      const int i = i0 + r + 8 * lh;
      const int j = j0 + s * 16 + ln;
      const size_t idx = base + (size_t)i * NSEQ + j;
      float tv = (need_t1 && j <= i) ? accT[s][r] : 0.0f;
      t1[idx] = f2bf(tv);
      float lv = 0.0f;
      if (need_la && j > i) lv = 1.0f / (1.0f + __expf(-accL[s][r]));
      la[idx] = f2bf(lv);
    }
  }
}

// ---------------------------------------------------------------------------
// Kernel 3: fused flash attention over causal k-tiles.
// Per (b,h, 64-row i-block): for kt = 0..i_blk/64:
//   Su tile via term1 @ lookahead^T with triangular K-bounds (software
//   pipelined: next fragments load while current WMMAs execute),
//   Sc tile via qc_s @ kc^T, score = Sc - silu(Su) (+causal mask),
//   online softmax in LDS, O = O*rescale + P @ vc.
// 8 waves: wave (i_sub = w>>1) x (k/dd subtile pair ks0 = (w&1)*2).
// ---------------------------------------------------------------------------
__global__ void __launch_bounds__(256)
castle_attn(const uint16_t* __restrict__ qc, const uint16_t* __restrict__ kc,
            const uint16_t* __restrict__ vcT, const uint16_t* __restrict__ t1,
            const uint16_t* __restrict__ la, uint16_t* __restrict__ out_h) {
  __shared__ float    S[64 * 66];
  __shared__ uint16_t P[64 * 72];
  __shared__ float    m_s[64], l_s[64], sc_s[64];

  const int bh     = blockIdx.y;
  const int i_blk0 = blockIdx.x * 64;
  const int tid    = threadIdx.x;
  const int wave   = tid >> 5;
  const int lane   = tid & 31;
  const int lh = lane >> 4, ln = lane & 15;
  const int i_sub = wave >> 1;           // 0..3 (16-row subtile)
  const int ks0   = (wave & 1) * 2;      // subtile pair for cols (k) / dd

  const uint16_t* qc_b  = qc  + (size_t)bh * NSEQ * DH;
  const uint16_t* kc_b  = kc  + (size_t)bh * NSEQ * DH;
  const uint16_t* vcT_b = vcT + (size_t)bh * DH * NSEQ;
  const uint16_t* t1_b  = t1  + (size_t)bh * NSEQ * NSEQ;
  const uint16_t* la_b  = la  + (size_t)bh * NSEQ * NSEQ;

  if (tid < 64) { m_s[tid] = -3.0e38f; l_s[tid] = 0.0f; }
  __syncthreads();

  v8f o0 = {}, o1 = {};
  const int rowA  = i_blk0 + i_sub * 16 + ln;
  const int ktMax = i_blk0 >> 6;
  const int jEnd  = i_blk0 + 64;

  for (int kt = 0; kt <= ktMax; ++kt) {
    const int k0col = kt * 64;
    const int c0 = k0col + ks0 * 16 + ln;        // lookahead/kc row (= score col)
    const int c1 = k0col + (ks0 + 1) * 16 + ln;

    // prefetch next k-tile's lookahead rows (global_prefetch_b8 path)
    if (kt < ktMax) {
      __builtin_prefetch(la_b + (size_t)(c0 + 64) * NSEQ + k0col, 0, 1);
      __builtin_prefetch(la_b + (size_t)(c1 + 64) * NSEQ + k0col, 0, 1);
    }

    // --- phase 1a: Su (triangular K range), register double-buffered -------
    v8f su0 = {}, su1 = {};
    {
      v16bf aC  = load_a_frag(t1_b, rowA, NSEQ, k0col, lane);
      v16bf b0C = load_b_frag(la_b, c0, NSEQ, k0col, lane);
      v16bf b1C = load_b_frag(la_b, c1, NSEQ, k0col, lane);
      for (int j0 = k0col; j0 + 32 < jEnd; j0 += 32) {   // trip count >= 1
        const v16bf aN  = load_a_frag(t1_b, rowA, NSEQ, j0 + 32, lane);
        const v16bf b0N = load_b_frag(la_b, c0, NSEQ, j0 + 32, lane);
        const v16bf b1N = load_b_frag(la_b, c1, NSEQ, j0 + 32, lane);
        su0 = wmma_bf16(aC, b0C, su0);
        su1 = wmma_bf16(aC, b1C, su1);
        aC = aN; b0C = b0N; b1C = b1N;
      }
      su0 = wmma_bf16(aC, b0C, su0);   // epilogue step
      su1 = wmma_bf16(aC, b1C, su1);
    }
    // --- phase 1b: Sc (K=64, loads hoisted) --------------------------------
    v8f sc0 = {}, sc1 = {};
    {
      const v16bf a0  = load_a_frag(qc_b, rowA, DH, 0, lane);
      const v16bf a1  = load_a_frag(qc_b, rowA, DH, 32, lane);
      const v16bf b00 = load_b_frag(kc_b, c0, DH, 0, lane);
      const v16bf b01 = load_b_frag(kc_b, c0, DH, 32, lane);
      const v16bf b10 = load_b_frag(kc_b, c1, DH, 0, lane);
      const v16bf b11 = load_b_frag(kc_b, c1, DH, 32, lane);
      sc0 = wmma_bf16(a0, b00, sc0);
      sc0 = wmma_bf16(a1, b01, sc0);
      sc1 = wmma_bf16(a0, b10, sc1);
      sc1 = wmma_bf16(a1, b11, sc1);
    }
#pragma unroll
    for (int r = 0; r < 8; ++r) {
      const int ir = i_sub * 16 + r + 8 * lh;   // local row 0..63
      const int ig = i_blk0 + ir;
      {
        const float su = su0[r];
        float sc = sc0[r] - su / (1.0f + __expf(-su));   // Sc - silu(Su)
        if (c0 - ln + (ln) > ig) sc = sc;                 // keep expression simple
        if (c0 > ig) sc = -3.0e38f;
        S[ir * 66 + ks0 * 16 + ln] = sc;
      }
      {
        const float su = su1[r];
        float sc = sc1[r] - su / (1.0f + __expf(-su));
        if (c1 > ig) sc = -3.0e38f;
        S[ir * 66 + (ks0 + 1) * 16 + ln] = sc;
      }
    }
    __syncthreads();
    // --- phase 2: per-row online softmax update ----------------------------
    if (tid < 64) {
      const float mold = m_s[tid];
      float mnew = mold;
#pragma unroll 4
      for (int c = 0; c < 64; ++c) mnew = fmaxf(mnew, S[tid * 66 + c]);
      float ssum = 0.0f;
#pragma unroll 4
      for (int c = 0; c < 64; ++c) {
        const float p = __expf(S[tid * 66 + c] - mnew);
        P[tid * 72 + c] = f2bf(p);
        ssum += p;
      }
      const float scl = __expf(mold - mnew);
      m_s[tid] = mnew;
      l_s[tid] = l_s[tid] * scl + ssum;
      sc_s[tid] = scl;
    }
    __syncthreads();
    // --- phase 3: rescale O, accumulate P @ vc (loads hoisted) -------------
#pragma unroll
    for (int r = 0; r < 8; ++r) {
      const float scl = sc_s[i_sub * 16 + r + 8 * lh];
      o0[r] *= scl;
      o1[r] *= scl;
    }
    {
      const int d0 = ks0 * 16 + ln, d1 = (ks0 + 1) * 16 + ln;
      const v16bf a0  = load_a_frag(P, i_sub * 16 + ln, 72, 0, lane);
      const v16bf a1  = load_a_frag(P, i_sub * 16 + ln, 72, 32, lane);
      const v16bf b00 = load_b_frag(vcT_b, d0, NSEQ, k0col, lane);
      const v16bf b01 = load_b_frag(vcT_b, d0, NSEQ, k0col + 32, lane);
      const v16bf b10 = load_b_frag(vcT_b, d1, NSEQ, k0col, lane);
      const v16bf b11 = load_b_frag(vcT_b, d1, NSEQ, k0col + 32, lane);
      o0 = wmma_bf16(a0, b00, o0);
      o0 = wmma_bf16(a1, b01, o0);
      o1 = wmma_bf16(a0, b10, o1);
      o1 = wmma_bf16(a1, b11, o1);
    }
    __syncthreads();
  }

  // --- epilogue: normalize, store bf16 [B, N, h*64] ------------------------
  const int b = bh >> 3, h = bh & 7;
#pragma unroll
  for (int r = 0; r < 8; ++r) {
    const int ir = i_sub * 16 + r + 8 * lh;
    const int ig = i_blk0 + ir;
    const float inv = 1.0f / l_s[ir];
    const size_t base = ((size_t)(b * NSEQ + ig)) * (NH * DH) + h * DH;
    out_h[base + ks0 * 16 + ln]       = f2bf(o0[r] * inv);
    out_h[base + (ks0 + 1) * 16 + ln] = f2bf(o1[r] * inv);
  }
}

// ---------------------------------------------------------------------------
// Kernel 4: output projection. out[4096,1024] = out_h @ W_out^T, f32 result.
// One wave = 32x64 tile, K=512, double-buffered.
// ---------------------------------------------------------------------------
__global__ void __launch_bounds__(128)
out_gemm(const uint16_t* __restrict__ oh, const uint16_t* __restrict__ wo,
         float* __restrict__ out) {
  const int wid  = (blockIdx.x * 128 + threadIdx.x) >> 5;
  const int lane = threadIdx.x & 31;
  const int lh = lane >> 4, ln = lane & 15;
  const int mt = wid >> 4, nt = wid & 15;
  const int m0 = mt * 32, n0 = nt * 64;

  v8f acc[2][4] = {};
  const int rA0 = m0 + ln, rA1 = m0 + 16 + ln;

  v16bf a0c = load_a_frag(oh, rA0, 512, 0, lane);
  v16bf a1c = load_a_frag(oh, rA1, 512, 0, lane);
  v16bf bc0 = load_b_frag(wo, n0 + 0 * 16 + ln, 512, 0, lane);
  v16bf bc1 = load_b_frag(wo, n0 + 1 * 16 + ln, 512, 0, lane);
  v16bf bc2 = load_b_frag(wo, n0 + 2 * 16 + ln, 512, 0, lane);
  v16bf bc3 = load_b_frag(wo, n0 + 3 * 16 + ln, 512, 0, lane);

  for (int k0 = 0; k0 < 512; k0 += 32) {
    v16bf a0n = {}, a1n = {}, bn0 = {}, bn1 = {}, bn2 = {}, bn3 = {};
    const int kn = k0 + 32;
    if (kn < 512) {
      a0n = load_a_frag(oh, rA0, 512, kn, lane);
      a1n = load_a_frag(oh, rA1, 512, kn, lane);
      bn0 = load_b_frag(wo, n0 + 0 * 16 + ln, 512, kn, lane);
      bn1 = load_b_frag(wo, n0 + 1 * 16 + ln, 512, kn, lane);
      bn2 = load_b_frag(wo, n0 + 2 * 16 + ln, 512, kn, lane);
      bn3 = load_b_frag(wo, n0 + 3 * 16 + ln, 512, kn, lane);
    }
    acc[0][0] = wmma_bf16(a0c, bc0, acc[0][0]);
    acc[0][1] = wmma_bf16(a0c, bc1, acc[0][1]);
    acc[0][2] = wmma_bf16(a0c, bc2, acc[0][2]);
    acc[0][3] = wmma_bf16(a0c, bc3, acc[0][3]);
    acc[1][0] = wmma_bf16(a1c, bc0, acc[1][0]);
    acc[1][1] = wmma_bf16(a1c, bc1, acc[1][1]);
    acc[1][2] = wmma_bf16(a1c, bc2, acc[1][2]);
    acc[1][3] = wmma_bf16(a1c, bc3, acc[1][3]);
    a0c = a0n; a1c = a1n; bc0 = bn0; bc1 = bn1; bc2 = bn2; bc3 = bn3;
  }
#pragma unroll
  for (int g = 0; g < 2; ++g) {
#pragma unroll
    for (int s = 0; s < 4; ++s) {
#pragma unroll
      for (int r = 0; r < 8; ++r) {
        out[(size_t)(m0 + g * 16 + r + 8 * lh) * 1024 + n0 + s * 16 + ln] =
            acc[g][s][r];
      }
    }
  }
}

// ---------------------------------------------------------------------------
extern "C" void kernel_launch(void* const* d_in, const int* in_sizes, int n_in,
                              void* d_out, int out_size, void* d_ws, size_t ws_size,
                              hipStream_t stream) {
  const float* x    = (const float*)d_in[0];
  const float* Wqkv = (const float*)d_in[1];
  const float* Wout = (const float*)d_in[2];
  float* out = (float*)d_out;

  uint16_t* ws  = (uint16_t*)d_ws;
  uint16_t* qkv = ws;                 // 6 slabs
  uint16_t* t1  = ws + T1_E;
  uint16_t* la  = ws + LA_E;
  uint16_t* oh  = ws + OH_E;
  uint16_t* xb  = ws + XB_E;
  uint16_t* wqb = ws + WQ_E;
  uint16_t* wob = ws + WO_E;

  const int nx = NB * NSEQ * 1024;
  const int nq = 3072 * 1024;
  const int no = 1024 * 512;
  cvt_kernel<<<(nx + 255) / 256, 256, 0, stream>>>(x, xb, nx);
  cvt_kernel<<<(nq + 255) / 256, 256, 0, stream>>>(Wqkv, wqb, nq);
  cvt_kernel<<<(no + 255) / 256, 256, 0, stream>>>(Wout, wob, no);

  // QKV GEMM: 128 m-tiles x 48 n-tiles = 6144 waves -> 1536 blocks of 128
  qkv_gemm<<<1536, 128, 0, stream>>>(xb, wqb, qkv);

  // term1/lookahead: 16 bh x 128 i-tiles x 32 j-tiles = 65536 waves
  tl_kernel<<<16384, 128, 0, stream>>>(qkv + 0 * SLAB_E,   // qu_s
                                       qkv + 1 * SLAB_E,   // ku
                                       qkv + 2 * SLAB_E,   // vu
                                       qkv + 3 * SLAB_E,   // qc_s
                                       t1, la);

  // fused flash: 32 i-blocks x 16 (b,h), 256 threads
  castle_attn<<<dim3(32, 16), 256, 0, stream>>>(qkv + 3 * SLAB_E,  // qc_s
                                                qkv + 4 * SLAB_E,  // kc
                                                qkv + 5 * SLAB_E,  // vcT
                                                t1, la, oh);

  // output projection: 128 m-tiles x 16 n-tiles = 2048 waves -> 512 blocks
  out_gemm<<<512, 128, 0, stream>>>(oh, wob, out);
}
